// GlobalPointer2_38903813767735
// MI455X (gfx1250) — matrix operand order repaired and checked by reading, not compiled, not run on hardware
//
#include <hip/hip_runtime.h>
#include <hip/hip_bf16.h>

#define HEADS 12
#define HEAD_SIZE 64
#define HIDDEN 768
#define BATCH 8
#define SEQ 512
#define NQK (HEADS * HEAD_SIZE * 2)   // 1536
#define MTOT (BATCH * SEQ)            // 4096
#define KSTEP 64
#define LDT 72                        // padded LDS row stride (144B, 16B-aligned frags)

typedef __attribute__((ext_vector_type(16))) __bf16 v16bf;
typedef __attribute__((ext_vector_type(8)))  __bf16 v8bf;
typedef __attribute__((ext_vector_type(4)))  __bf16 v4bf;
typedef __attribute__((ext_vector_type(8)))  float  v8f;
typedef __attribute__((ext_vector_type(4)))  float  v4f;

// fp32 -> bf16 round-to-nearest-even (keeps fp32 exponent range)
__device__ __forceinline__ __bf16 f2bf(float f) {
  union { float f; unsigned u; } v; v.f = f;
  unsigned r = v.u + 0x7FFFu + ((v.u >> 16) & 1u);
  unsigned short h = (unsigned short)(r >> 16);
  __bf16 o; __builtin_memcpy(&o, &h, 2);
  return o;
}

__device__ __forceinline__ v16bf cat16(v8bf lo, v8bf hi) {
  v16bf o;
#pragma unroll
  for (int j = 0; j < 8; ++j) { o[j] = lo[j]; o[8 + j] = hi[j]; }
  return o;
}

// ---------------------------------------------------------------- convert X (f32 -> bf16)
__global__ __launch_bounds__(256)
void cvt_f32_bf16(const float* __restrict__ src, __bf16* __restrict__ dst, int n4) {
  int t = blockIdx.x * blockDim.x + threadIdx.x;
  if (t >= n4) return;
  v4f x = *(const v4f*)(src + (size_t)t * 4);
  v4bf y;
  y[0] = f2bf(x[0]); y[1] = f2bf(x[1]); y[2] = f2bf(x[2]); y[3] = f2bf(x[3]);
  *(v4bf*)(dst + (size_t)t * 4) = y;
}

// ---------------------------------------------------------------- convert+transpose W
// W[768][1536] f32 -> Wt[1536][768] bf16 (one-time tiled transpose via LDS)
__global__ __launch_bounds__(256)
void cvt_transpose_W(const float* __restrict__ W, __bf16* __restrict__ Wt) {
  __shared__ __bf16 tile[32][36];           // padded
  const int t  = threadIdx.x;
  const int kb = blockIdx.y * 32;           // 768/32 = 24
  const int nb = blockIdx.x * 32;           // 1536/32 = 48
  {
    const int kk  = t >> 3;                 // 0..31
    const int nn4 = (t & 7) * 4;            // 0..28
    v4f x = *(const v4f*)(W + (size_t)(kb + kk) * NQK + nb + nn4);
#pragma unroll
    for (int j = 0; j < 4; ++j) tile[kk][nn4 + j] = f2bf(x[j]);
  }
  __syncthreads();
  {
    const int nn  = t >> 3;
    const int kk4 = (t & 7) * 4;
    v4bf y;
#pragma unroll
    for (int j = 0; j < 4; ++j) y[j] = tile[kk4 + j][nn];
    *(v4bf*)(Wt + (size_t)(nb + nn) * HIDDEN + kb + kk4) = y;
  }
}

// ---------------------------------------------------------------- GEMM1 + bias + RoPE
// out(4096x1536) = X @ W + b, RoPE, store bf16 Q/K [b,h,s,d].
// Block: 256 thr (8 waves, 4M x 2N), tile 64M x 128N, K-step 64, double-buffered LDS.
__global__ __launch_bounds__(256)
void qkv_rope_kernel(const __bf16* __restrict__ X, const __bf16* __restrict__ Wt,
                     const float* __restrict__ bias,
                     __bf16* __restrict__ Qb, __bf16* __restrict__ Kb) {
  __shared__ __bf16 sA[2][64 * LDT];
  __shared__ __bf16 sB[2][128 * LDT];

  const int t    = threadIdx.x;
  const int lane = t & 31;
  const int wid  = t >> 5;
  const int wm   = wid & 3;       // 4 waves along M
  const int wn   = wid >> 2;      // 2 waves along N
  const int r    = lane & 15;
  const int hg   = lane >> 4;     // lane half per CDNA5 WMMA layout
  const int Mbase = blockIdx.y * 64;
  const int Nbase = blockIdx.x * 128;

  const int srow = t >> 2;               // 0..63
  const int scol = (t & 3) * 16;         // 0,16,32,48

  v8f acc[4] = {};
  v8bf a0, a1, b0, b1, b2, b3;

  auto gload = [&](int kb) {
    const __bf16* xr = X  + (size_t)(Mbase + srow) * HIDDEN + kb + scol;
    const __bf16* w0 = Wt + (size_t)(Nbase + srow) * HIDDEN + kb + scol;
    const __bf16* w1 = Wt + (size_t)(Nbase + 64 + srow) * HIDDEN + kb + scol;
    a0 = *(const v8bf*)(xr);     a1 = *(const v8bf*)(xr + 8);
    b0 = *(const v8bf*)(w0);     b1 = *(const v8bf*)(w0 + 8);
    b2 = *(const v8bf*)(w1);     b3 = *(const v8bf*)(w1 + 8);
  };
  auto lstore = [&](int buf) {
    *(v8bf*)(&sA[buf][srow * LDT + scol])            = a0;
    *(v8bf*)(&sA[buf][srow * LDT + scol + 8])        = a1;
    *(v8bf*)(&sB[buf][srow * LDT + scol])            = b0;
    *(v8bf*)(&sB[buf][srow * LDT + scol + 8])        = b1;
    *(v8bf*)(&sB[buf][(64 + srow) * LDT + scol])     = b2;
    *(v8bf*)(&sB[buf][(64 + srow) * LDT + scol + 8]) = b3;
  };

  gload(0);
  lstore(0);
  __syncthreads();

  constexpr int NIT = HIDDEN / KSTEP;   // 12
  int cur = 0;
  for (int it = 0; it < NIT; ++it) {
    if (it + 1 < NIT) gload((it + 1) * KSTEP);

#pragma unroll
    for (int k2 = 0; k2 < 2; ++k2) {
      // batch all fragment loads of this k-slice, THEN issue the WMMA burst,
      // so ds_load latency drains across the whole group instead of per-WMMA.
      const __bf16* ap = &sA[cur][(wm * 16 + r) * LDT + k2 * 32 + hg * 16];
      v16bf a = cat16(*(const v8bf*)ap, *(const v8bf*)(ap + 8));
      v16bf bb[4];
#pragma unroll
      for (int sub = 0; sub < 4; ++sub) {
        const __bf16* bp = &sB[cur][(wn * 64 + sub * 16 + r) * LDT + k2 * 32 + hg * 16];
        bb[sub] = cat16(*(const v8bf*)bp, *(const v8bf*)(bp + 8));
      }
#pragma unroll
      for (int sub = 0; sub < 4; ++sub)
        acc[sub] = __builtin_amdgcn_wmma_f32_16x16x32_bf16(
            false, a, false, bb[sub], (short)0, acc[sub], false, false);
    }

    if (it + 1 < NIT) {
      lstore(cur ^ 1);
      __syncthreads();
      cur ^= 1;
    }
  }

  // epilogue: bias + RoPE + scatter to Q/K (bf16)
#pragma unroll
  for (int sub = 0; sub < 4; ++sub) {
    const int n    = Nbase + wn * 64 + sub * 16 + r;   // C layout: col = lane%16
    const int h    = n >> 7;
    const int d    = n & 127;
    const int is_q = (d < HEAD_SIZE);
    const int dd   = d & (HEAD_SIZE - 1);
    const int ip   = dd >> 1;
    const float invf = __expf(-0.2878231366f * (float)ip);  // 10000^(-2i/64)
    const float bn   = bias[n];
    __bf16* dst = is_q ? Qb : Kb;
#pragma unroll
    for (int i = 0; i < 8; ++i) {
      const int m    = Mbase + wm * 16 + hg * 8 + i;   // C layout: row = hg*8 + vgpr
      const int s    = m & (SEQ - 1);
      const int bidx = m >> 9;
      float v = acc[sub][i] + bn;
      float p = __shfl_xor(v, 1, 32);          // pair col n^1 lives in lane^1
      float ang = invf * (float)s;
      float cs = __cosf(ang), sn = __sinf(ang);
      float rot = (dd & 1) ? p : -p;
      float o = v * cs + rot * sn;
      dst[(((size_t)bidx * HEADS + h) * SEQ + s) * HEAD_SIZE + dd] = f2bf(o);
    }
  }
}

// ---------------------------------------------------------------- GEMM2 + masks + scale
// logits[b,h,m,n] = sum_d Q[m,d]*K[n,d]; pad/tril mask; /8.
// Block: 8 waves (4M x 2N), tile 64M x 128N, K=64: all fragments loaded up front,
// then 8 back-to-back WMMAs.
__global__ __launch_bounds__(256)
void attn_logits_kernel(const __bf16* __restrict__ Qb, const __bf16* __restrict__ Kb,
                        const float* __restrict__ mask, float* __restrict__ out) {
  const int t    = threadIdx.x;
  const int lane = t & 31;
  const int wid  = t >> 5;
  const int wm   = wid & 3;
  const int wn   = wid >> 2;
  const int r    = lane & 15;
  const int hg   = lane >> 4;
  const int bh   = blockIdx.y;            // 0..95
  const int bidx = bh / HEADS;
  const int mt = blockIdx.x >> 2, nt = blockIdx.x & 3;   // 8 x 4 tiles
  const int Mbase = mt * 64, Nbase = nt * 128;
  const __bf16* Q = Qb + (size_t)bh * SEQ * HEAD_SIZE;
  const __bf16* K = Kb + (size_t)bh * SEQ * HEAD_SIZE;

  // load all fragments first (pure global_load_b128), then the WMMA burst
  v16bf a[2], bb[2][4];
#pragma unroll
  for (int k2 = 0; k2 < 2; ++k2) {
    const int kb = k2 * 32;
    const __bf16* ap = Q + (size_t)(Mbase + wm * 16 + r) * HEAD_SIZE + kb + hg * 16;
    a[k2] = cat16(*(const v8bf*)ap, *(const v8bf*)(ap + 8));
#pragma unroll
    for (int sub = 0; sub < 4; ++sub) {
      const int n = Nbase + wn * 64 + sub * 16 + r;
      const __bf16* bp = K + (size_t)n * HEAD_SIZE + kb + hg * 16;   // contiguous 32B
      bb[k2][sub] = cat16(*(const v8bf*)bp, *(const v8bf*)(bp + 8));
    }
  }

  v8f acc[4] = {};
#pragma unroll
  for (int k2 = 0; k2 < 2; ++k2)
#pragma unroll
    for (int sub = 0; sub < 4; ++sub)
      acc[sub] = __builtin_amdgcn_wmma_f32_16x16x32_bf16(
          false, a[k2], false, bb[k2][sub], (short)0, acc[sub], false, false);

  const float* mrow = mask + (size_t)bidx * SEQ;
#pragma unroll
  for (int sub = 0; sub < 4; ++sub) {
    const int n = Nbase + wn * 64 + sub * 16 + r;
    const float pad  = mrow[n];
    const float negp = (1.0f - pad) * 1.0e12f;
#pragma unroll
    for (int i = 0; i < 8; ++i) {
      const int m = Mbase + wm * 16 + hg * 8 + i;
      float v = acc[sub][i] * pad - negp;
      v -= (n < m) ? 1.0e12f : 0.0f;          // tril(ones, -1) mask
      out[((size_t)bh * SEQ + m) * SEQ + n] = v * 0.125f;   // / sqrt(64)
    }
  }
}

// ---------------------------------------------------------------- launch
extern "C" void kernel_launch(void* const* d_in, const int* in_sizes, int n_in,
                              void* d_out, int out_size, void* d_ws, size_t ws_size,
                              hipStream_t stream) {
  const float* inputs = (const float*)d_in[0];   // [8,512,768]
  const float* mask   = (const float*)d_in[1];   // [8,512]
  const float* W      = (const float*)d_in[2];   // [768,1536]
  const float* bias   = (const float*)d_in[3];   // [1536]
  float* out = (float*)d_out;                    // [8,12,512,512]

  __bf16* Xbf = (__bf16*)d_ws;                           // 4096*768
  __bf16* Wt  = Xbf + (size_t)MTOT * HIDDEN;             // 1536*768 (transposed)
  __bf16* Qb  = Wt  + (size_t)NQK * HIDDEN;              // 96*512*64
  __bf16* Kb  = Qb  + (size_t)BATCH * HEADS * SEQ * HEAD_SIZE;

  {
    int n4 = (MTOT * HIDDEN) / 4;
    cvt_f32_bf16<<<(n4 + 255) / 256, 256, 0, stream>>>(inputs, Xbf, n4);
  }
  {
    dim3 grid(NQK / 32, HIDDEN / 32);   // 48 x 24
    cvt_transpose_W<<<grid, 256, 0, stream>>>(W, Wt);
  }
  {
    dim3 grid(NQK / 128, MTOT / 64);    // 12 x 64
    qkv_rope_kernel<<<grid, 256, 0, stream>>>(Xbf, Wt, bias, Qb, Kb);
  }
  {
    dim3 grid(32, BATCH * HEADS);       // (8 mtiles * 4 ntiles) x 96
    attn_logits_kernel<<<grid, 256, 0, stream>>>(Qb, Kb, mask, out);
  }
}
